// VQVAEEncoder_39273180955354
// MI455X (gfx1250) — matrix operand order repaired and checked by reading, not compiled, not run on hardware
//
#include <hip/hip_runtime.h>

typedef __attribute__((ext_vector_type(16))) _Float16 v16h;
typedef __attribute__((ext_vector_type(8)))  _Float16 v8h;
typedef __attribute__((ext_vector_type(8)))  float    v8f;
typedef __attribute__((ext_vector_type(4)))  unsigned int v4u;
typedef __attribute__((ext_vector_type(8)))  int v8i_t;
typedef __attribute__((ext_vector_type(4)))  int v4i_t;

// ---------------------------------------------------------------------------
// Tensor Data Mover: 1-D DMA of a contiguous block (global f16 -> LDS).
// D# per CDNA5 ISA ch.8 (group0: count/lds/global/type; group1: data_size=8B,
// 1-D tensor+tile dims). Low 32 bits of a flat pointer to a __shared__ object
// are the LDS byte offset. 6-argument builtin form on this toolchain.
// ---------------------------------------------------------------------------
__device__ __forceinline__ void tdm_load_tile(const _Float16* gsrc, _Float16* ldst,
                                              int nbytes)
{
#if __has_builtin(__builtin_amdgcn_tensor_load_to_lds)
    unsigned lds_off = (unsigned)(unsigned long long)(uintptr_t)ldst;
    unsigned long long ga = (unsigned long long)(uintptr_t)gsrc;
    unsigned n8 = ((unsigned)nbytes) >> 3;            // 8-byte elements
    v4u g0;
    g0.x = 1u;                                        // count=1, user mode
    g0.y = lds_off;                                   // lds_addr
    g0.z = (unsigned)(ga & 0xffffffffu);              // global_addr[31:0]
    g0.w = (unsigned)((ga >> 32) & 0x01ffffffu)       // global_addr[56:32]
         | (2u << 30);                                // type=2 ("image")
    v8i_t g1;
    g1[0] = (int)(3u << 16);                          // data_size=3 (8B)
    g1[1] = (int)((n8 & 0xffffu) << 16);              // tensor_dim0[15:0]
    g1[2] = (int)((n8 >> 16) | (1u << 16));           // tensor_dim0[31:16], tensor_dim1=1
    g1[3] = (int)((n8 & 0xffffu) << 16);              // tile_dim0 = n8
    g1[4] = 1;                                        // tile_dim1 = 1
    g1[5] = (int)n8;                                  // tensor_dim0_stride[31:0]
    g1[6] = 0;
    g1[7] = 0;
    v4i_t z4 = {0, 0, 0, 0};
    v8i_t z8 = {0, 0, 0, 0, 0, 0, 0, 0};
    __builtin_amdgcn_tensor_load_to_lds(g0, g1, z4, z4, z8, 0);
#else
    const unsigned long long* s = (const unsigned long long*)gsrc;
    unsigned long long* d = (unsigned long long*)ldst;
    int n = nbytes >> 3;
    for (int i = threadIdx.x & 31; i < n; i += 32) d[i] = s[i];
#endif
}

__device__ __forceinline__ void tdm_wait()
{
#if __has_builtin(__builtin_amdgcn_s_wait_tensorcnt)
    __builtin_amdgcn_s_wait_tensorcnt(0);
#endif
}

// ---------------------------------------------------------------------------
// Implicit-GEMM conv1d (conv-transpose = even/odd phase decomposition) using
// V_WMMA_F32_16X16X32_F16. One workgroup = OPAD/16 waves; wave w computes the
// 16(o) x 16(t) tile at o0 = 16w. Each wave TDM-DMAs its own weight slice;
// all threads cooperatively stage ONE shared im2col tile (no redundant
// activation traffic across o-tiles).
// mode: 0 = relu+store, 1 = store, 2 = loss (compare vs ref, atomic SSE)
// ---------------------------------------------------------------------------
template<int I, int KW, int KPAD, int OPAD>
__global__ __launch_bounds__(32 * (OPAD / 16))
void vqvae_conv_wmma(const float* __restrict__ in, const _Float16* __restrict__ wgt,
                     const float* __restrict__ bias, float* __restrict__ out,
                     const float* __restrict__ ref, float* __restrict__ sse_acc,
                     int stride, int offset, int T_in, int O_real, int T_outlen,
                     int ostride, int phase, int mode)
{
    constexpr int NW = OPAD / 16;
    constexpr int KTOT = I * KW;
    __shared__ __align__(16) _Float16 ldsW[OPAD * KPAD];
    __shared__ __align__(16) _Float16 ldsB[16 * KPAD];

    const int tid  = threadIdx.x;
    const int lane = tid & 31;
    const int wv   = tid >> 5;
    const int t0 = blockIdx.x * 16;
    const int o0 = wv * 16;
    const int b  = blockIdx.z;

    // Per-wave async DMA of this wave's 16-row weight slice into its LDS
    // partition; overlaps with the cooperative VALU im2col staging below.
    tdm_load_tile(wgt + (size_t)o0 * KPAD, &ldsW[o0 * KPAD], 16 * KPAD * 2);

    // Stage ONE im2col activation tile [16 cols x KPAD], f32 -> f16, shared
    // by all waves (o-tiles).
    for (int idx = tid; idx < 16 * KPAD; idx += 32 * NW) {
        int n = idx / KPAD, kk = idx - n * KPAD;
        float v = 0.0f;
        if (kk < KTOT) {
            int i = kk / KW, k = kk - i * KW;
            int src = stride * (t0 + n) + offset + k;
            if (src >= 0 && src < T_in)
                v = in[((size_t)b * I + i) * T_in + src];
        }
        ldsB[idx] = (_Float16)v;
    }
    tdm_wait();          // wave's own weight DMA complete (only reads own slice)
    __syncthreads();     // ldsB complete across all waves

    const int half = lane >> 4;   // K-half selector per CDNA5 WMMA layout
    const int l16  = lane & 15;   // A row (m) / B col (n)

    v8f c = {};
    #pragma unroll
    for (int ks = 0; ks < KPAD; ks += 32) {
        const _Float16* rw = &ldsW[(o0 + l16) * KPAD + ks];
        const _Float16* rb = &ldsB[l16 * KPAD + ks];
        v8h alo = *(const v8h*)(rw + half * 8);
        v8h ahi = *(const v8h*)(rw + 16 + half * 8);
        v8h blo = *(const v8h*)(rb + half * 8);
        v8h bhi = *(const v8h*)(rb + 16 + half * 8);
        v16h a  = __builtin_shufflevector(alo, ahi, 0,1,2,3,4,5,6,7,8,9,10,11,12,13,14,15);
        v16h bv = __builtin_shufflevector(blo, bhi, 0,1,2,3,4,5,6,7,8,9,10,11,12,13,14,15);
        c = __builtin_amdgcn_wmma_f32_16x16x32_f16(false, a, false, bv,
                                                   (short)0, c, false, false);
    }

    // Epilogue: D layout — lane l16 = column n, VGPR r = row m = half*8 + r.
    float lsse = 0.0f;
    #pragma unroll
    for (int r = 0; r < 8; ++r) {
        int o = o0 + half * 8 + r;
        int t = t0 + l16;
        if (o < O_real) {
            float v = c[r] + bias[o];
            if (mode == 0) v = fmaxf(v, 0.0f);
            size_t oidx = ((size_t)b * O_real + o) * T_outlen
                        + (size_t)ostride * t + phase;
            if (mode == 2) {
                float d = v - ref[oidx];
                lsse += d * d;
            } else {
                out[oidx] = v;
            }
        }
    }
    if (mode == 2) {
        lsse += __shfl_xor(lsse, 16);
        lsse += __shfl_xor(lsse, 8);
        lsse += __shfl_xor(lsse, 4);
        lsse += __shfl_xor(lsse, 2);
        lsse += __shfl_xor(lsse, 1);
        if (lane == 0) atomicAdd(sse_acc, lsse);
    }
}

// ---------------------------------------------------------------------------
// Vector quantizer: NW waves per block share one TDM-loaded copy of the f16
// codebook in LDS. Each wave handles 16 z_e vectors: -2*z.c via WMMA against
// all 512 codes, branchless argmin of ||c||^2 - 2 z.c, gather z_q, accumulate
// SSE + conditioning sums.
// ---------------------------------------------------------------------------
template<int NW>
__global__ __launch_bounds__(32 * NW)
void vqvae_quantize_wmma(const float* __restrict__ ze, const _Float16* __restrict__ cbh,
                         const float* __restrict__ cbf, const float* __restrict__ cnorm,
                         float* __restrict__ zq, float* __restrict__ cond_acc,
                         float* __restrict__ sseq_acc, int T3)
{
    __shared__ __align__(16) _Float16 ldsCB[512 * 64];   // 64 KB f16 codebook
    __shared__ float    ldsCn[512];
    __shared__ __align__(16) _Float16 ldsZ[NW][16 * 64];
    __shared__ float    ldsZf[NW][16 * 64];
    __shared__ int      ldsIdx[NW][16];

    const int tid  = threadIdx.x;
    const int lane = tid & 31;
    const int wv   = tid >> 5;
    const int tilesPerB = T3 / 16;
    const int gtile = blockIdx.x * NW + wv;
    const int b  = gtile / tilesPerB;
    const int tt = gtile % tilesPerB;
    const int t0 = tt * 16;

    // Each wave DMAs its 1/NW slice of the codebook into LDS.
    constexpr int CSL = 512 / NW;
    tdm_load_tile(cbh + (size_t)wv * CSL * 64, &ldsCB[wv * CSL * 64], CSL * 64 * 2);

    for (int i = tid; i < 512; i += 32 * NW) ldsCn[i] = cnorm[i];

    // Stage this wave's z tile: [v][d] (f32 copy kept for exact loss).
    for (int idx = lane; idx < 16 * 64; idx += 32) {
        int v = idx >> 6, d = idx & 63;
        float z = ze[((size_t)b * 64 + d) * T3 + t0 + v];
        ldsZf[wv][idx] = z;
        ldsZ[wv][idx]  = (_Float16)z;
    }
    tdm_wait();          // own slice done...
    __syncthreads();     // ...barrier makes ALL slices + z tiles visible

    const int half = lane >> 4;
    const int l16  = lane & 15;

    float bestd = 3.4e38f;
    int   besti = 0;
    for (int ct = 0; ct < 32; ++ct) {
        v8f c = {};
        #pragma unroll
        for (int ks = 0; ks < 64; ks += 32) {
            const _Float16* rw = &ldsCB[(ct * 16 + l16) * 64 + ks];
            const _Float16* rb = &ldsZ[wv][l16 * 64 + ks];
            v8h alo = *(const v8h*)(rw + half * 8);
            v8h ahi = *(const v8h*)(rw + 16 + half * 8);
            v8h blo = *(const v8h*)(rb + half * 8);
            v8h bhi = *(const v8h*)(rb + 16 + half * 8);
            v16h a  = __builtin_shufflevector(alo, ahi, 0,1,2,3,4,5,6,7,8,9,10,11,12,13,14,15);
            v16h bv = __builtin_shufflevector(blo, bhi, 0,1,2,3,4,5,6,7,8,9,10,11,12,13,14,15);
            c = __builtin_amdgcn_wmma_f32_16x16x32_f16(false, a, false, bv,
                                                       (short)0, c, false, false);
        }
        // ||c||^2 for this tile's 8 codes: two aligned float4 LDS loads.
        const float4 cnA = *(const float4*)&ldsCn[ct * 16 + half * 8];
        const float4 cnB = *(const float4*)&ldsCn[ct * 16 + half * 8 + 4];
        float cn[8] = {cnA.x, cnA.y, cnA.z, cnA.w, cnB.x, cnB.y, cnB.z, cnB.w};
        // Branchless argmin; codes ascend per lane so strict < keeps first min.
        #pragma unroll
        for (int r = 0; r < 8; ++r) {
            int code = ct * 16 + half * 8 + r;
            float d2 = cn[r] - 2.0f * c[r];
            bool tk = d2 < bestd;
            besti = tk ? code : besti;
            bestd = tk ? d2 : bestd;
        }
    }
    // Lanes L and L^16 hold the same column n=l16; combine with index tiebreak.
    float od = __shfl_xor(bestd, 16);
    int   oi = __shfl_xor(besti, 16);
    bool tk = (od < bestd) || (od == bestd && oi < besti);
    besti = tk ? oi : besti;
    bestd = tk ? od : bestd;
    if (lane < 16) ldsIdx[wv][lane] = besti;
    __syncthreads();

    float lsse = 0.0f;
    float condp0 = 0.0f, condp1 = 0.0f;
    for (int idx = lane; idx < 16 * 64; idx += 32) {
        int v = idx >> 6, d = idx & 63;     // d alternates {lane, lane+32}
        float q = cbf[(size_t)ldsIdx[wv][v] * 64 + d];
        float z = ldsZf[wv][idx];
        float df = q - z;
        lsse += df * df;
        zq[((size_t)b * 64 + d) * T3 + t0 + v] = q;
        if (((idx >> 5) & 1) == 0) condp0 += q; else condp1 += q;
    }
    atomicAdd(&cond_acc[b * 64 + lane],      condp0);
    atomicAdd(&cond_acc[b * 64 + lane + 32], condp1);
    lsse += __shfl_xor(lsse, 16);
    lsse += __shfl_xor(lsse, 8);
    lsse += __shfl_xor(lsse, 4);
    lsse += __shfl_xor(lsse, 2);
    lsse += __shfl_xor(lsse, 1);
    if (lane == 0) atomicAdd(sseq_acc, lsse);
}

// ---------------------------------------------------------------------------
// Helper kernels.
// ---------------------------------------------------------------------------
__global__ void vqvae_pack_conv_w(const float* __restrict__ w, _Float16* __restrict__ out,
                                  int O_real, int I, int KW, int Kpad, int O_pad)
{
    int idx = blockIdx.x * blockDim.x + threadIdx.x;
    int total = O_pad * Kpad;
    if (idx >= total) return;
    int o = idx / Kpad, kk = idx - o * Kpad;
    float v = 0.0f;
    if (o < O_real && kk < I * KW) v = w[(size_t)o * I * KW + kk];
    out[idx] = (_Float16)v;
}

// ConvTranspose1d(stride=2,pad=1,k=4): out_even[u] taps {3,1} at in[u-1],in[u];
// out_odd[u] taps {2,0} at in[u],in[u+1]. dw is [I][O][4].
__global__ void vqvae_pack_deconv_w(const float* __restrict__ dw,
                                    _Float16* __restrict__ outE, _Float16* __restrict__ outO,
                                    int I, int O_real, int O_pad)
{
    int Kpad = 2 * I;
    int idx = blockIdx.x * blockDim.x + threadIdx.x;
    int total = O_pad * Kpad;
    if (idx >= total) return;
    int o = idx / Kpad, kk = idx - o * Kpad;
    int i = kk >> 1, k2 = kk & 1;
    float ve = 0.0f, vo = 0.0f;
    if (o < O_real) {
        ve = dw[((size_t)i * O_real + o) * 4 + (k2 ? 1 : 3)];
        vo = dw[((size_t)i * O_real + o) * 4 + (k2 ? 0 : 2)];
    }
    outE[idx] = (_Float16)ve;
    outO[idx] = (_Float16)vo;
}

__global__ void vqvae_pack_codebook(const float* __restrict__ cb,
                                    _Float16* __restrict__ cbh, float* __restrict__ cnorm)
{
    int c = blockIdx.x * blockDim.x + threadIdx.x;
    if (c >= 512) return;
    float s = 0.0f;
    for (int d = 0; d < 64; ++d) {
        float v = cb[c * 64 + d];
        s += v * v;
        cbh[c * 64 + d] = (_Float16)v;
    }
    cnorm[c] = s;
}

__global__ void vqvae_zero(float* __restrict__ p, int n)
{
    int i = blockIdx.x * blockDim.x + threadIdx.x;
    if (i < n) p[i] = 0.0f;
}

__global__ void vqvae_finalize(const float* __restrict__ acc, float* __restrict__ out, int T3)
{
    int i = blockIdx.x * blockDim.x + threadIdx.x;
    if (i < 1024) out[i] = acc[i] / (float)T3;     // conditioning mean over T
    if (i == 0) {
        float vq = acc[1024] / (float)(16 * 64 * 4096);
        out[1024] = vq;                            // vq_loss
        out[1025] = vq;                            // commitment_loss (same fwd value)
        out[1026] = acc[1025] / (float)(16 * 8 * 32768);  // recon_loss
    }
}

// ---------------------------------------------------------------------------
extern "C" void kernel_launch(void* const* d_in, const int* in_sizes, int n_in,
                              void* d_out, int out_size, void* d_ws, size_t ws_size,
                              hipStream_t stream)
{
    (void)in_sizes; (void)n_in; (void)out_size; (void)ws_size;
    const float* x   = (const float*)d_in[0];
    const float* w1  = (const float*)d_in[1];
    const float* b1  = (const float*)d_in[2];
    const float* w2  = (const float*)d_in[3];
    const float* b2  = (const float*)d_in[4];
    const float* w3  = (const float*)d_in[5];
    const float* b3  = (const float*)d_in[6];
    const float* cb  = (const float*)d_in[7];
    const float* dw1 = (const float*)d_in[8];
    const float* db1 = (const float*)d_in[9];
    const float* dw2 = (const float*)d_in[10];
    const float* db2 = (const float*)d_in[11];
    const float* dw3 = (const float*)d_in[12];
    const float* db3 = (const float*)d_in[13];

    char* ws = (char*)d_ws;
    float* bufA = (float*)(ws + 0);                    // z1 [16,64,16384] then r1 [16,128,8192]
    float* bufB = (float*)(ws + 67108864);             // z2 [16,128,8192] then r2 [16,64,16384]
    float* z_e  = (float*)(ws + 134217728);            // [16,64,4096]
    float* z_q  = (float*)(ws + 150994944);            // [16,64,4096]
    _Float16* pk = (_Float16*)(ws + 167772160);
    _Float16* pw1  = pk + 0;        // 64*64
    _Float16* pw2  = pk + 4096;     // 128*320
    _Float16* pw3  = pk + 45056;    // 64*384
    _Float16* pd1e = pk + 69632;    // 128*128
    _Float16* pd1o = pk + 86016;
    _Float16* pd2e = pk + 102400;   // 64*256
    _Float16* pd2o = pk + 118784;
    _Float16* pd3e = pk + 135168;   // 16*128
    _Float16* pd3o = pk + 137216;
    _Float16* pcb  = pk + 139264;   // 512*64
    float* cnorm = (float*)(ws + 168116224);           // 512
    float* acc   = (float*)(ws + 168118272);           // [0..1023]=cond, [1024]=sse_q, [1025]=sse_r

    // --- prep ---
    vqvae_pack_conv_w<<<16, 256, 0, stream>>>(w1, pw1, 64, 8, 7, 64, 64);
    vqvae_pack_conv_w<<<160, 256, 0, stream>>>(w2, pw2, 128, 64, 5, 320, 128);
    vqvae_pack_conv_w<<<96, 256, 0, stream>>>(w3, pw3, 64, 128, 3, 384, 64);
    vqvae_pack_deconv_w<<<64, 256, 0, stream>>>(dw1, pd1e, pd1o, 64, 128, 128);
    vqvae_pack_deconv_w<<<64, 256, 0, stream>>>(dw2, pd2e, pd2o, 128, 64, 64);
    vqvae_pack_deconv_w<<<8, 256, 0, stream>>>(dw3, pd3e, pd3o, 64, 8, 16);
    vqvae_pack_codebook<<<2, 256, 0, stream>>>(cb, pcb, cnorm);
    vqvae_zero<<<5, 256, 0, stream>>>(acc, 1026);

    // --- encoder ---
    vqvae_conv_wmma<8, 7, 64, 64><<<dim3(1024, 1, 16), 128, 0, stream>>>(
        x, pw1, b1, bufA, nullptr, nullptr, 2, -3, 32768, 64, 16384, 1, 0, 0);
    vqvae_conv_wmma<64, 5, 320, 128><<<dim3(512, 1, 16), 256, 0, stream>>>(
        bufA, pw2, b2, bufB, nullptr, nullptr, 2, -2, 16384, 128, 8192, 1, 0, 0);
    vqvae_conv_wmma<128, 3, 384, 64><<<dim3(256, 1, 16), 128, 0, stream>>>(
        bufB, pw3, b3, z_e, nullptr, nullptr, 2, -1, 8192, 64, 4096, 1, 0, 1);

    // --- quantize: 512 blocks x 8 waves, codebook shared in LDS via TDM ---
    vqvae_quantize_wmma<8><<<dim3(512), 256, 0, stream>>>(z_e, pcb, cb, cnorm, z_q,
        acc, acc + 1024, 4096);

    // --- decoder (z_q_st == z_q numerically) ---
    vqvae_conv_wmma<64, 2, 128, 128><<<dim3(256, 1, 16), 256, 0, stream>>>(
        z_q, pd1e, db1, bufA, nullptr, nullptr, 1, -1, 4096, 128, 8192, 2, 0, 0);
    vqvae_conv_wmma<64, 2, 128, 128><<<dim3(256, 1, 16), 256, 0, stream>>>(
        z_q, pd1o, db1, bufA, nullptr, nullptr, 1, 0, 4096, 128, 8192, 2, 1, 0);
    vqvae_conv_wmma<128, 2, 256, 64><<<dim3(512, 1, 16), 128, 0, stream>>>(
        bufA, pd2e, db2, bufB, nullptr, nullptr, 1, -1, 8192, 64, 16384, 2, 0, 0);
    vqvae_conv_wmma<128, 2, 256, 64><<<dim3(512, 1, 16), 128, 0, stream>>>(
        bufA, pd2o, db2, bufB, nullptr, nullptr, 1, 0, 8192, 64, 16384, 2, 1, 0);
    vqvae_conv_wmma<64, 2, 128, 16><<<dim3(1024, 1, 16), 32, 0, stream>>>(
        bufB, pd3e, db3, bufA, x, acc + 1025, 1, -1, 16384, 8, 32768, 2, 0, 2);
    vqvae_conv_wmma<64, 2, 128, 16><<<dim3(1024, 1, 16), 32, 0, stream>>>(
        bufB, pd3o, db3, bufA, x, acc + 1025, 1, 0, 16384, 8, 32768, 2, 1, 2);

    // --- outputs ---
    vqvae_finalize<<<5, 256, 0, stream>>>(acc, (float*)d_out, 4096);
}
